// DSQGAttention_82729660056300
// MI455X (gfx1250) — compile-verified, matrix operand order checked by reading, not run on hardware
//
#include <hip/hip_runtime.h>
#include <hip/hip_bf16.h>

// ---------------------------------------------------------------------------
// DSQG attention for MI455X (gfx1250, wave32, WMMA).
// Pipeline:
//   1) cvt x -> bf16 ; cvt+transpose w_qkv/w_gate/w_out -> bf16 [N][K]
//   2) WMMA GEMM (bf16 -> f32 acc):
//        mode 0: qkv, scatter q(*0.125)/k/v -> bf16 [B,H,N,HD]
//        mode 1: gate = sigmoid(x@w_gate + b) -> f32 [BN,D]
//        mode 2: out  = act@w_out + b        -> f32 d_out
//   3) attention: wave32 per (b,h,n), 44 dyadic taps, softmax, v-gather,
//      gate multiply, store bf16 act for final GEMM.
// ---------------------------------------------------------------------------

typedef __attribute__((ext_vector_type(16))) __bf16 v16bf;
typedef __attribute__((ext_vector_type(8)))  float   v8f;
typedef __attribute__((ext_vector_type(8)))  unsigned int v8u;

#define BATCH 4
#define NSEQ  4096
#define DMODEL 1024
#define NHEAD 16
#define HDIM  64
#define NTAPS 44

static __device__ __forceinline__ unsigned short f2bf(float f) {
  unsigned int u = __float_as_uint(f);
  u += 0x7FFFu + ((u >> 16) & 1u);        // round-to-nearest-even
  return (unsigned short)(u >> 16);
}
static __device__ __forceinline__ float bf2f(unsigned int h) {
  return __uint_as_float(h << 16);
}

// ---------------------------------------------------------------------------
// Conversions
// ---------------------------------------------------------------------------
__global__ void cvt_f32_to_bf16(const float* __restrict__ in,
                                unsigned short* __restrict__ out, size_t n) {
  size_t i = (size_t)blockIdx.x * blockDim.x + threadIdx.x;
  if (i < n) out[i] = f2bf(in[i]);
}

// in [K][NC] row-major f32  ->  out [NC][K] row-major bf16 (transposed)
__global__ void cvt_transpose_bf16(const float* __restrict__ in,
                                   unsigned short* __restrict__ out,
                                   int K, int NC) {
  size_t i = (size_t)blockIdx.x * blockDim.x + threadIdx.x;
  size_t total = (size_t)K * NC;
  if (i < total) {
    int k = (int)(i / NC);
    int n = (int)(i % NC);
    out[(size_t)n * K + k] = f2bf(in[i]);
  }
}

// ---------------------------------------------------------------------------
// Fragment loader: 16-bit A/B fragment (16 rows x 32 K) from row-major bf16,
// per CDNA5 ISA 7.12.2:
//   lanes 0-15 : row = lane,     K pairs {0,2,4,6} and {16,18,20,22}
//   lanes16-31 : row = lane-16,  K pairs shifted by +8
// Each VGPR = one contiguous bf16 pair -> single dword load.
// ---------------------------------------------------------------------------
static __device__ __forceinline__ v16bf
load_frag16(const unsigned short* __restrict__ p, int ld, int row_base,
            int k_base, int lane) {
  int r = row_base + (lane & 15);
  int khalf = (lane & 16) ? 8 : 0;
  const unsigned short* rp = p + (size_t)r * ld + k_base + khalf;
  v8u u;
#pragma unroll
  for (int v = 0; v < 8; ++v) {
    int ko = (v < 4) ? (2 * v) : (8 + 2 * v);  // 0,2,4,6,16,18,20,22
    u[v] = *(const unsigned int*)(rp + ko);
  }
  union { v8u ui; v16bf bf; } cvt;
  cvt.ui = u;
  return cvt.bf;
}

// ---------------------------------------------------------------------------
// Generic WMMA GEMM: C[M,NC] = A[M,K](bf16) * Bt[NC,K](bf16)^T  (+epilogue)
// block = 256 threads = 8 waves; block tile 128(M) x 64(N); wave = 16x64.
// ---------------------------------------------------------------------------
__global__ __launch_bounds__(256) void gemm_wmma_bf16(
    const unsigned short* __restrict__ A,
    const unsigned short* __restrict__ Bt,
    int K, int NC, int mode,
    const float* __restrict__ bias,
    unsigned short* __restrict__ qT,  // mode 0
    unsigned short* __restrict__ kT,  // mode 0
    unsigned short* __restrict__ vT,  // mode 0
    float* __restrict__ gateOut,      // mode 1
    float* __restrict__ outF32)       // mode 2
{
  const int lane = threadIdx.x & 31;
  const int wave = threadIdx.x >> 5;
  const int mBase = blockIdx.x * 128 + wave * 16;
  const int nBase = blockIdx.y * 64;

  v8f acc0 = {}, acc1 = {}, acc2 = {}, acc3 = {};

  for (int k = 0; k < K; k += 32) {
    v16bf a = load_frag16(A, K, mBase, k, lane);
    v16bf b0 = load_frag16(Bt, K, nBase +  0, k, lane);
    v16bf b1 = load_frag16(Bt, K, nBase + 16, k, lane);
    v16bf b2 = load_frag16(Bt, K, nBase + 32, k, lane);
    v16bf b3 = load_frag16(Bt, K, nBase + 48, k, lane);
    acc0 = __builtin_amdgcn_wmma_f32_16x16x32_bf16(false, a, false, b0,
                                                   (short)0, acc0, false, false);
    acc1 = __builtin_amdgcn_wmma_f32_16x16x32_bf16(false, a, false, b1,
                                                   (short)0, acc1, false, false);
    acc2 = __builtin_amdgcn_wmma_f32_16x16x32_bf16(false, a, false, b2,
                                                   (short)0, acc2, false, false);
    acc3 = __builtin_amdgcn_wmma_f32_16x16x32_bf16(false, a, false, b3,
                                                   (short)0, acc3, false, false);
  }

  // C/D layout (7.12.2): VGPR r, lanes0-15 -> (M=r, N=lane); lanes16-31 -> (M=8+r)
  const int mOff = (lane & 16) ? 8 : 0;
  const int nLane = lane & 15;
  v8f accs[4] = {acc0, acc1, acc2, acc3};

#pragma unroll
  for (int t = 0; t < 4; ++t) {
    int n = nBase + t * 16 + nLane;
    float bn = bias[n];
#pragma unroll
    for (int r = 0; r < 8; ++r) {
      int m = mBase + mOff + r;
      float val = accs[t][r] + bn;
      if (mode == 0) {
        int b  = m >> 12;           // / NSEQ
        int nn = m & (NSEQ - 1);
        int which = n >> 10;        // 0=q 1=k 2=v
        int d = n & (DMODEL - 1);
        int h = d >> 6, hd = d & (HDIM - 1);
        size_t idx = ((((size_t)b * NHEAD + h) * NSEQ) + nn) * HDIM + hd;
        if (which == 0)      qT[idx] = f2bf(val * 0.125f);  // pre-scaled by HD^-0.5
        else if (which == 1) kT[idx] = f2bf(val);
        else                 vT[idx] = f2bf(val);
      } else if (mode == 1) {
        gateOut[(size_t)m * NC + n] = 1.0f / (1.0f + __expf(-val));
      } else {
        outF32[(size_t)m * NC + n] = val;
      }
    }
  }
}

// ---------------------------------------------------------------------------
// Dyadic attention: one wave32 per (b,h,n). Each lane owns 2 channels of HD=64.
// score_i = q . k[n-off_i]  (shifted-in zeros when n<off -> score = pos_bias),
// softmax over 44 taps, out = sum attn_i * v[n-off_i], then * gate, -> bf16 act.
// ---------------------------------------------------------------------------
__global__ __launch_bounds__(256) void attn_dyadic(
    const unsigned short* __restrict__ qT,
    const unsigned short* __restrict__ kT,
    const unsigned short* __restrict__ vT,
    const float* __restrict__ gate,      // [B*N, D] f32
    const float* __restrict__ pos_bias,  // [44, H] f32
    unsigned short* __restrict__ act)    // [B*N, D] bf16
{
  const int lane = threadIdx.x & 31;
  const int wave = threadIdx.x >> 5;
  const size_t pos = (size_t)blockIdx.x * 8 + wave;  // (b*H + h)*N + n
  const int n = (int)(pos & (NSEQ - 1));
  const size_t bh = pos >> 12;                       // / NSEQ
  const int h = (int)(bh & (NHEAD - 1));
  const size_t b = bh >> 4;                          // / NHEAD
  const size_t rowBase = pos * HDIM;

  float q0, q1;
  {
    unsigned int u = *(const unsigned int*)(qT + rowBase + 2 * lane);
    q0 = bf2f(u & 0xFFFFu); q1 = bf2f(u >> 16);
  }

  float sc[NTAPS];
#pragma unroll
  for (int i = 0; i < NTAPS; ++i) {
    const int off = (i & 3) << (i >> 2);     // (1<<j)*tau
    float s = 0.0f;
    if (n >= off) {
      unsigned int u =
          *(const unsigned int*)(kT + rowBase - (size_t)off * HDIM + 2 * lane);
      s = q0 * bf2f(u & 0xFFFFu) + q1 * bf2f(u >> 16);
    }
#pragma unroll
    for (int d = 16; d > 0; d >>= 1) s += __shfl_xor(s, d, 32);
    sc[i] = s + pos_bias[i * NHEAD + h];
  }

  // softmax over the 44 taps (replicated across lanes)
  float mx = sc[0];
#pragma unroll
  for (int i = 1; i < NTAPS; ++i) mx = fmaxf(mx, sc[i]);
  float sum = 0.0f;
#pragma unroll
  for (int i = 0; i < NTAPS; ++i) { sc[i] = __expf(sc[i] - mx); sum += sc[i]; }
  const float inv = 1.0f / sum;

  float o0 = 0.0f, o1 = 0.0f;
#pragma unroll
  for (int i = 0; i < NTAPS; ++i) {
    const int off = (i & 3) << (i >> 2);
    if (n >= off) {
      unsigned int u =
          *(const unsigned int*)(vT + rowBase - (size_t)off * HDIM + 2 * lane);
      const float w = sc[i] * inv;
      o0 += w * bf2f(u & 0xFFFFu);
      o1 += w * bf2f(u >> 16);
    }
  }

  const size_t rowX = b * NSEQ + n;            // token row in [B*N]
  const int d0 = h * HDIM + 2 * lane;
  const float g0 = gate[rowX * DMODEL + d0];
  const float g1 = gate[rowX * DMODEL + d0 + 1];
  const unsigned int packed =
      ((unsigned int)f2bf(o1 * g1) << 16) | (unsigned int)f2bf(o0 * g0);
  *(unsigned int*)(act + rowX * DMODEL + d0) = packed;
}

// ---------------------------------------------------------------------------
// Host-side launch
// ---------------------------------------------------------------------------
extern "C" void kernel_launch(void* const* d_in, const int* in_sizes, int n_in,
                              void* d_out, int out_size, void* d_ws,
                              size_t ws_size, hipStream_t stream) {
  const float* x        = (const float*)d_in[0];  // [B,N,D]
  const float* w_qkv    = (const float*)d_in[1];  // [D,3D]
  const float* b_qkv    = (const float*)d_in[2];  // [3D]
  const float* w_out    = (const float*)d_in[3];  // [D,D]
  const float* b_out    = (const float*)d_in[4];  // [D]
  const float* w_gate   = (const float*)d_in[5];  // [D,D]
  const float* b_gate   = (const float*)d_in[6];  // [D]
  const float* pos_bias = (const float*)d_in[7];  // [44,H]
  float* out = (float*)d_out;                     // [B,N,D]

  const int M = BATCH * NSEQ;          // 16384
  const int K = DMODEL;                // 1024
  const size_t BN_D   = (size_t)M * DMODEL;
  const size_t QKV_E  = (size_t)BATCH * NHEAD * NSEQ * HDIM;  // 16.8M

  // workspace carve-out (256B aligned)
  char* ws = (char*)d_ws;
  size_t off = 0;
  auto carve = [&](size_t bytes) -> void* {
    off = (off + 255) & ~(size_t)255;
    void* p = ws + off;
    off += bytes;
    return p;
  };
  unsigned short* xb     = (unsigned short*)carve(BN_D * 2);             // x bf16
  unsigned short* wqkvT  = (unsigned short*)carve((size_t)3 * K * K * 2);
  unsigned short* wgateT = (unsigned short*)carve((size_t)K * K * 2);
  unsigned short* woutT  = (unsigned short*)carve((size_t)K * K * 2);
  unsigned short* qT     = (unsigned short*)carve(QKV_E * 2);
  unsigned short* kT     = (unsigned short*)carve(QKV_E * 2);
  unsigned short* vT     = (unsigned short*)carve(QKV_E * 2);
  float*          gateB  = (float*)carve(BN_D * 4);
  unsigned short* act    = (unsigned short*)carve(BN_D * 2);

  // 1) conversions
  {
    size_t n = BN_D;
    cvt_f32_to_bf16<<<(unsigned)((n + 255) / 256), 256, 0, stream>>>(x, xb, n);
  }
  {
    size_t n = (size_t)K * (3 * DMODEL);
    cvt_transpose_bf16<<<(unsigned)((n + 255) / 256), 256, 0, stream>>>(
        w_qkv, wqkvT, K, 3 * DMODEL);
  }
  {
    size_t n = (size_t)K * DMODEL;
    cvt_transpose_bf16<<<(unsigned)((n + 255) / 256), 256, 0, stream>>>(
        w_gate, wgateT, K, DMODEL);
    cvt_transpose_bf16<<<(unsigned)((n + 255) / 256), 256, 0, stream>>>(
        w_out, woutT, K, DMODEL);
  }

  // 2) qkv GEMM (mode 0): [16384,1024] x [1024,3072]
  {
    dim3 grid(M / 128, (3 * DMODEL) / 64);
    gemm_wmma_bf16<<<grid, 256, 0, stream>>>(xb, wqkvT, K, 3 * DMODEL, 0,
                                             b_qkv, qT, kT, vT, nullptr,
                                             nullptr);
  }
  // 3) gate GEMM (mode 1): sigmoid(x@w_gate + b_gate)
  {
    dim3 grid(M / 128, DMODEL / 64);
    gemm_wmma_bf16<<<grid, 256, 0, stream>>>(xb, wgateT, K, DMODEL, 1, b_gate,
                                             nullptr, nullptr, nullptr, gateB,
                                             nullptr);
  }
  // 4) dyadic attention + gating -> bf16 activations
  {
    unsigned blocks = (unsigned)(((size_t)BATCH * NHEAD * NSEQ) / 8);  // 32768
    attn_dyadic<<<blocks, 256, 0, stream>>>(qT, kT, vT, gateB, pos_bias, act);
  }
  // 5) output GEMM (mode 2): act @ w_out + b_out -> f32 d_out
  {
    dim3 grid(M / 128, DMODEL / 64);
    gemm_wmma_bf16<<<grid, 256, 0, stream>>>(act, woutT, K, DMODEL, 2, b_out,
                                             nullptr, nullptr, nullptr, nullptr,
                                             out);
  }
}
